// ResidualGATLayer_81235011437200
// MI455X (gfx1250) — matrix operand order repaired and checked by reading, not compiled, not run on hardware
//
#include <hip/hip_runtime.h>

// ---------------- problem constants ----------------
#define NN      50000
#define IN_CH   128
#define HID     32
#define HEADS   3
#define OUT_CH  128
#define EDGES   800000LL
#define ETOT    (EDGES + (long long)NN)   // + self loops
#define NEG_SLOPE 0.2f
#define BN_EPS  1e-5f

// ---------------- WMMA vector types ----------------
typedef __attribute__((ext_vector_type(16))) __bf16          v16bf;
typedef __attribute__((ext_vector_type(16))) unsigned short  v16u;
typedef __attribute__((ext_vector_type(8)))  unsigned short  v8u;
typedef __attribute__((ext_vector_type(8)))  float           v8f;

__device__ __forceinline__ unsigned short f2bf(float f) {
  unsigned u = __float_as_uint(f);
  u += 0x7FFFu + ((u >> 16) & 1u);          // round-to-nearest-even
  return (unsigned short)(u >> 16);
}

__device__ __forceinline__ void atomicMaxF(float* addr, float val) {
  int* ai = (int*)addr;
  int cur = __float_as_int(*addr);
  while (__int_as_float(cur) < val) {
    int prev = atomicCAS(ai, cur, __float_as_int(val));
    if (prev == cur) break;
    cur = prev;
  }
}

__device__ __forceinline__ void edge_ends(const long long* __restrict__ ei,
                                          long long e, int& s, int& d) {
  if (e < EDGES) { s = (int)ei[e]; d = (int)ei[EDGES + e]; }
  else           { s = d = (int)(e - EDGES); }
}

// ---------------- utility kernels ----------------
__global__ void fill_f32(float* __restrict__ p, float v, long long n) {
  long long i = (long long)blockIdx.x * blockDim.x + threadIdx.x;
  if (i < n) p[i] = v;
}

__global__ void f32_to_bf16(const float* __restrict__ in,
                            unsigned short* __restrict__ out, long long n) {
  long long i = (long long)blockIdx.x * blockDim.x + threadIdx.x;
  if (i < n) out[i] = f2bf(in[i]);
}

// add per-column bias while converting (reference adds bias AFTER aggregation)
__global__ void f32_to_bf16_bias(const float* __restrict__ in,
                                 const float* __restrict__ bias, int ch,
                                 unsigned short* __restrict__ out, long long n) {
  long long i = (long long)blockIdx.x * blockDim.x + threadIdx.x;
  if (i < n) out[i] = f2bf(in[i] + bias[i % ch]);
}

__global__ void add_bias(float* __restrict__ p, const float* __restrict__ bias,
                         int ch, long long n) {
  long long i = (long long)blockIdx.x * blockDim.x + threadIdx.x;
  if (i < n) p[i] += bias[i % ch];
}

// ---------------- B-matrix pre-pack for WMMA ----------------
// Layout (this code's contiguous k = 16h + i, same as A; 16-bit B 32x16): lane n (0-15) = column, lanes 16-31 hold
// K=16..31.  Per lane 16 contiguous bf16 so GEMM can use b128 loads.
// Bp[(ntile*ktiles + kt)*32 + lane][i]  <- B[kbase + i][col]
__global__ void pack_B(const float* __restrict__ B, unsigned short* __restrict__ Bp,
                       int K, int Nd) {
  int ktiles = K >> 5, ntiles = Nd >> 4;
  int total = ntiles * ktiles * 32;
  int idx = blockIdx.x * blockDim.x + threadIdx.x;
  if (idx >= total) return;
  int lane  = idx & 31;
  int kt    = (idx >> 5) % ktiles;
  int ntile = (idx >> 5) / ktiles;
  int col   = (ntile << 4) + (lane & 15);
  int kbase = (kt << 5) + ((lane < 16) ? 0 : 16);
  unsigned short* dst = Bp + (size_t)idx * 16;
#pragma unroll
  for (int i = 0; i < 16; ++i) dst[i] = f2bf(B[(size_t)(kbase + i) * Nd + col]);
}

__device__ __forceinline__ v16bf frag_from(const unsigned short* p0,
                                           const unsigned short* p1) {
  v8u lo = *(const v8u*)p0;
  v8u hi = *(const v8u*)p1;
  v16u u;
#pragma unroll
  for (int i = 0; i < 8; ++i) { u[i] = lo[i]; u[8 + i] = hi[i]; }
  return __builtin_bit_cast(v16bf, u);
}

// ---------------- WMMA GEMM: C[MxNd] = A[MxK](bf16) * B(packed bf16) ----------
// one wave per 16x32 output strip (two 16x16 n-tiles share each A fragment);
// compile-time KTILES/NTILES -> fully unrolled v_wmma_f32_16x16x32_bf16 body
template <int KTILES, int NTILES>
__global__ __launch_bounds__(256)
void gemm_bf16_wmma(const unsigned short* __restrict__ A,
                    const unsigned short* __restrict__ Bp,
                    const float* __restrict__ bias,
                    float* __restrict__ C, int M) {
  constexpr int K  = KTILES * 32;
  constexpr int Nd = NTILES * 16;
  constexpr int NPAIRS = NTILES / 2;           // 96->3, 128->4 (exact)
  const int lane = threadIdx.x & 31;
  const int wave = (blockIdx.x * blockDim.x + threadIdx.x) >> 5;
  const int mtiles = M >> 4;
  if (wave >= mtiles * NPAIRS) return;         // whole-wave exit: EXEC all-1 at WMMA
  const int mtile = wave / NPAIRS;
  const int np    = wave - mtile * NPAIRS;

  const int row = (mtile << 4) + (lane & 15);  // A: lane = M row (16-bit A layout)
  const int kA  = (lane < 16) ? 0 : 16;
  const unsigned short* Abase = A + (size_t)row * K + kA;
  const unsigned short* B0 = Bp + ((size_t)(np * 2)     * KTILES * 32 + lane) * 16;
  const unsigned short* B1 = Bp + ((size_t)(np * 2 + 1) * KTILES * 32 + lane) * 16;

  v8f acc0 = {}, acc1 = {};
#pragma unroll
  for (int kt = 0; kt < KTILES; ++kt) {
    const unsigned short* Arow = Abase + (kt << 5);
    __builtin_prefetch(Arow + 32, 0, 1);       // -> global_prefetch_b8
    v16bf av  = frag_from(Arow, Arow + 8);
    v16bf bv0 = frag_from(B0 + (size_t)kt * 512, B0 + (size_t)kt * 512 + 8);
    v16bf bv1 = frag_from(B1 + (size_t)kt * 512, B1 + (size_t)kt * 512 + 8);
    acc0 = __builtin_amdgcn_wmma_f32_16x16x32_bf16(
        false, av, false, bv0, (short)0, acc0, false, false);
    acc1 = __builtin_amdgcn_wmma_f32_16x16x32_bf16(
        false, av, false, bv1, (short)0, acc1, false, false);
  }
  const int col0  = (np << 5) + (lane & 15);
  const float bb0 = bias ? bias[col0] : 0.0f;
  const float bb1 = bias ? bias[col0 + 16] : 0.0f;
  const int mbase = (mtile << 4) + ((lane >> 4) << 3);   // C/D layout: VGPR r = row
#pragma unroll
  for (int r = 0; r < 8; ++r) {
    C[(size_t)(mbase + r) * Nd + col0]      = acc0[r] + bb0;
    C[(size_t)(mbase + r) * Nd + col0 + 16] = acc1[r] + bb1;
  }
}

// ---------------- GAT pieces ----------------
__global__ void attn_scores(const float* __restrict__ h,
                            const float* __restrict__ asrc,
                            const float* __restrict__ adst,
                            float* __restrict__ as_, float* __restrict__ ad_,
                            int H, int C) {
  int i = blockIdx.x * blockDim.x + threadIdx.x;
  if (i >= NN) return;
  for (int hh = 0; hh < H; ++hh) {
    float s = 0.f, d = 0.f;
    const float* hp = h + (size_t)i * H * C + hh * C;
    for (int c = 0; c < C; ++c) {
      float v = hp[c];
      s += v * asrc[hh * C + c];
      d += v * adst[hh * C + c];
    }
    as_[i * H + hh] = s;
    ad_[i * H + hh] = d;
  }
}

__global__ void edge_max(const long long* __restrict__ ei,
                         const float* __restrict__ as_, const float* __restrict__ ad_,
                         float* __restrict__ m, int H) {
  long long e = (long long)blockIdx.x * blockDim.x + threadIdx.x;
  if (e >= ETOT) return;
  int s, d; edge_ends(ei, e, s, d);
  for (int hh = 0; hh < H; ++hh) {
    float v = as_[(size_t)s * H + hh] + ad_[(size_t)d * H + hh];
    v = v > 0.f ? v : NEG_SLOPE * v;
    atomicMaxF(&m[(size_t)d * H + hh], v);
  }
}

__global__ void edge_expsum(const long long* __restrict__ ei,
                            const float* __restrict__ as_, const float* __restrict__ ad_,
                            const float* __restrict__ m, float* __restrict__ expe,
                            float* __restrict__ denom, int H) {
  long long e = (long long)blockIdx.x * blockDim.x + threadIdx.x;
  if (e >= ETOT) return;
  int s, d; edge_ends(ei, e, s, d);
  for (int hh = 0; hh < H; ++hh) {
    float v = as_[(size_t)s * H + hh] + ad_[(size_t)d * H + hh];
    v = v > 0.f ? v : NEG_SLOPE * v;
    float ee = __expf(v - m[(size_t)d * H + hh]);
    expe[(size_t)e * H + hh] = ee;
    atomicAdd(&denom[(size_t)d * H + hh], ee);
  }
}

// one wave32 per edge; lanes stride channels
__global__ __launch_bounds__(256)
void edge_agg(const long long* __restrict__ ei,
              const float* __restrict__ expe, const float* __restrict__ denom,
              const float* __restrict__ h, float* __restrict__ out,
              int H, int C) {
  int lane = threadIdx.x & 31;
  long long e = ((long long)blockIdx.x * blockDim.x + threadIdx.x) >> 5;
  if (e >= ETOT) return;
  int s, d; edge_ends(ei, e, s, d);
  const int CH = H * C;
  for (int c = lane; c < CH; c += 32) {
    int hh = c / C;
    float a = expe[(size_t)e * H + hh] / (denom[(size_t)d * H + hh] + 1e-16f);
    atomicAdd(&out[(size_t)d * CH + c], a * h[(size_t)s * CH + c]);
  }
}

// ---------------- BatchNorm ----------------
__global__ void bn_stats(const float* __restrict__ out2,
                         float* __restrict__ sums, float* __restrict__ sumsq) {
  __shared__ float ls[OUT_CH], lq[OUT_CH];
  int c = threadIdx.x & (OUT_CH - 1);
  int half = threadIdx.x >> 7;  // 0 or 1 with 256 threads
  float s = 0.f, q = 0.f;
  for (int r = blockIdx.x * 2 + half; r < NN; r += gridDim.x * 2) {
    float v = out2[(size_t)r * OUT_CH + c];
    s += v; q += v * v;
  }
  if (half == 1) { ls[c] = s; lq[c] = q; }
  __syncthreads();
  if (half == 0) {
    s += ls[c]; q += lq[c];
    atomicAdd(&sums[c], s);
    atomicAdd(&sumsq[c], q);
  }
}

__global__ void bn_relu_res(float* __restrict__ out2,
                            const float* __restrict__ sums, const float* __restrict__ sumsq,
                            const float* __restrict__ gamma, const float* __restrict__ beta,
                            const float* __restrict__ resid) {
  long long i = (long long)blockIdx.x * blockDim.x + threadIdx.x;
  if (i >= (long long)NN * OUT_CH) return;
  int c = (int)(i & (OUT_CH - 1));
  const float inv = 1.0f / (float)NN;
  float mu  = sums[c] * inv;
  float var = sumsq[c] * inv - mu * mu;
  float y = (out2[i] - mu) * rsqrtf(var + BN_EPS) * gamma[c] + beta[c];
  y = fmaxf(y, 0.0f);
  out2[i] = y + resid[i];
}

// ---------------- host launcher ----------------
extern "C" void kernel_launch(void* const* d_in, const int* in_sizes, int n_in,
                              void* d_out, int out_size, void* d_ws, size_t ws_size,
                              hipStream_t stream) {
  (void)in_sizes; (void)n_in; (void)out_size; (void)ws_size;
  const float*     x    = (const float*)d_in[0];
  const long long* ei   = (const long long*)d_in[1];   // int64 per reference
  const float*     W1   = (const float*)d_in[2];
  const float*     aS1  = (const float*)d_in[3];
  const float*     aD1  = (const float*)d_in[4];
  const float*     b1   = (const float*)d_in[5];
  const float*     W2   = (const float*)d_in[6];
  const float*     aS2  = (const float*)d_in[7];
  const float*     aD2  = (const float*)d_in[8];
  const float*     b2   = (const float*)d_in[9];
  const float*     gam  = (const float*)d_in[10];
  const float*     bet  = (const float*)d_in[11];
  const float*     Wres = (const float*)d_in[12];
  const float*     bres = (const float*)d_in[13];
  float* out = (float*)d_out;

  char* wp = (char*)d_ws;
  auto alloc = [&](size_t bytes) -> void* {
    void* p = (void*)wp;
    wp += (bytes + 255) & ~(size_t)255;
    return p;
  };
  const int CH1 = HEADS * HID;  // 96
  unsigned short* xb    = (unsigned short*)alloc((size_t)NN * IN_CH * 2);
  unsigned short* W1p   = (unsigned short*)alloc((size_t)IN_CH * CH1 * 2);
  unsigned short* W2p   = (unsigned short*)alloc((size_t)CH1 * OUT_CH * 2);
  unsigned short* Wrp   = (unsigned short*)alloc((size_t)IN_CH * OUT_CH * 2);
  float* h1     = (float*)alloc((size_t)NN * CH1 * 4);
  float* resid  = (float*)alloc((size_t)NN * OUT_CH * 4);
  float* as1    = (float*)alloc((size_t)NN * HEADS * 4);
  float* ad1    = (float*)alloc((size_t)NN * HEADS * 4);
  float* m1     = (float*)alloc((size_t)NN * HEADS * 4);
  float* dn1    = (float*)alloc((size_t)NN * HEADS * 4);
  float* expe1  = (float*)alloc((size_t)ETOT * HEADS * 4);
  float* out1   = (float*)alloc((size_t)NN * CH1 * 4);
  float* h2     = (float*)alloc((size_t)NN * OUT_CH * 4);
  float* bnsum  = (float*)alloc(OUT_CH * 4);
  float* bnsq   = (float*)alloc(OUT_CH * 4);
  // aliases (earlier buffers dead by the time these are live)
  unsigned short* out1b = xb;        // N*96 bf16 <= N*128 bf16
  float* expe2 = expe1;              // ETOT*1 <= ETOT*3
  float* as2 = as1; float* ad2 = ad1; float* m2 = m1; float* dn2 = dn1;

  const int T = 256;
  auto blk1d = [](long long n, int t) { return (unsigned)((n + t - 1) / t); };

  // --- prep: bf16 x, packed weights ---
  f32_to_bf16<<<blk1d((long long)NN * IN_CH, T), T, 0, stream>>>(x, xb, (long long)NN * IN_CH);
  pack_B<<<blk1d((CH1 / 16) * (IN_CH / 32) * 32, T), T, 0, stream>>>(W1, W1p, IN_CH, CH1);
  pack_B<<<blk1d((OUT_CH / 16) * (CH1 / 32) * 32, T), T, 0, stream>>>(W2, W2p, CH1, OUT_CH);
  pack_B<<<blk1d((OUT_CH / 16) * (IN_CH / 32) * 32, T), T, 0, stream>>>(Wres, Wrp, IN_CH, OUT_CH);

  // --- GEMMs: h1 = x@W1 (no bias; bias joins after aggregation), resid = x@Wres + bres ---
  {
    long long waves = (long long)(NN / 16) * (CH1 / 32);   // n-tile pairs
    gemm_bf16_wmma<4, 6><<<blk1d(waves * 32, T), T, 0, stream>>>(xb, W1p, nullptr, h1, NN);
    waves = (long long)(NN / 16) * (OUT_CH / 32);
    gemm_bf16_wmma<4, 8><<<blk1d(waves * 32, T), T, 0, stream>>>(xb, Wrp, bres, resid, NN);
  }

  // --- layer 1 attention ---
  attn_scores<<<blk1d(NN, T), T, 0, stream>>>(h1, aS1, aD1, as1, ad1, HEADS, HID);
  fill_f32<<<blk1d((long long)NN * HEADS, T), T, 0, stream>>>(m1, -1e30f, (long long)NN * HEADS);
  fill_f32<<<blk1d((long long)NN * HEADS, T), T, 0, stream>>>(dn1, 0.f, (long long)NN * HEADS);
  fill_f32<<<blk1d((long long)NN * CH1, T), T, 0, stream>>>(out1, 0.f, (long long)NN * CH1);
  edge_max<<<blk1d(ETOT, T), T, 0, stream>>>(ei, as1, ad1, m1, HEADS);
  edge_expsum<<<blk1d(ETOT, T), T, 0, stream>>>(ei, as1, ad1, m1, expe1, dn1, HEADS);
  edge_agg<<<blk1d(ETOT * 32, T), T, 0, stream>>>(ei, expe1, dn1, h1, out1, HEADS, HID);

  // --- layer 2: (out1 + b1) -> bf16 -> h2 = @W2 ---
  f32_to_bf16_bias<<<blk1d((long long)NN * CH1, T), T, 0, stream>>>(out1, b1, CH1, out1b, (long long)NN * CH1);
  {
    long long waves = (long long)(NN / 16) * (OUT_CH / 32);
    gemm_bf16_wmma<3, 8><<<blk1d(waves * 32, T), T, 0, stream>>>(out1b, W2p, nullptr, h2, NN);
  }
  attn_scores<<<blk1d(NN, T), T, 0, stream>>>(h2, aS2, aD2, as2, ad2, 1, OUT_CH);
  fill_f32<<<blk1d(NN, T), T, 0, stream>>>(m2, -1e30f, NN);
  fill_f32<<<blk1d(NN, T), T, 0, stream>>>(dn2, 0.f, NN);
  fill_f32<<<blk1d((long long)NN * OUT_CH, T), T, 0, stream>>>(out, 0.f, (long long)NN * OUT_CH);
  edge_max<<<blk1d(ETOT, T), T, 0, stream>>>(ei, as2, ad2, m2, 1);
  edge_expsum<<<blk1d(ETOT, T), T, 0, stream>>>(ei, as2, ad2, m2, expe2, dn2, 1);
  edge_agg<<<blk1d(ETOT * 32, T), T, 0, stream>>>(ei, expe2, dn2, h2, out, 1, OUT_CH);
  add_bias<<<blk1d((long long)NN * OUT_CH, T), T, 0, stream>>>(out, b2, OUT_CH, (long long)NN * OUT_CH);

  // --- batchnorm (training stats) + relu + residual, in place on d_out ---
  fill_f32<<<1, OUT_CH, 0, stream>>>(bnsum, 0.f, OUT_CH);
  fill_f32<<<1, OUT_CH, 0, stream>>>(bnsq, 0.f, OUT_CH);
  bn_stats<<<512, 256, 0, stream>>>(out, bnsum, bnsq);
  bn_relu_res<<<blk1d((long long)NN * OUT_CH, T), T, 0, stream>>>(out, bnsum, bnsq, gam, bet, resid);
}
